// SoftGatedChannelStack_13073880449226
// MI455X (gfx1250) — matrix operand order repaired and checked by reading, not compiled, not run on hardware
//
#include <hip/hip_runtime.h>

#define F_DIM 256
#define N_EXP 8
#define D_DIM 128
#define BM    32

typedef __attribute__((ext_vector_type(16))) __bf16 v16bf;
typedef __attribute__((ext_vector_type(8)))  float  v8f;

union Frag {
    v16bf v;
    uint4 q[2];
};

__device__ __forceinline__ unsigned short f2bf(float f) {
    unsigned u = __float_as_uint(f);
    unsigned r = u + 0x7FFFu + ((u >> 16) & 1u);   // round-to-nearest-even
    return (unsigned short)(r >> 16);
}
__device__ __forceinline__ unsigned pack2(float a, float b) {
    return (unsigned)f2bf(a) | ((unsigned)f2bf(b) << 16);
}

// ---------------------------------------------------------------------------
// Kernel 1: gating.  One thread per sample; Wg/Wn staged in LDS (broadcast).
// Produces: G (to d_out tail), and per-(b,n) {gate, slot} with the slot
// bijection: selected experts -> slots 0..k-1 (gate=G), unselected experts
// -> slots k..7 (gate=0).  Every slot written exactly once by kernel 2.
// ---------------------------------------------------------------------------
__global__ void __launch_bounds__(256)
gate_kernel(const float* __restrict__ x,
            const float* __restrict__ noise,
            const float* __restrict__ Wg_w, const float* __restrict__ Wg_b,
            const float* __restrict__ Wn_w, const float* __restrict__ Wn_b,
            float* __restrict__ G_out,
            float* __restrict__ gate_ws,
            int*   __restrict__ slot_ws)
{
    __shared__ __align__(16) float wg[N_EXP * F_DIM];
    __shared__ __align__(16) float wn[N_EXP * F_DIM];
    const int t = threadIdx.x;
    for (int i = t; i < N_EXP * F_DIM; i += 256) {
        wg[i] = Wg_w[i];
        wn[i] = Wn_w[i];
    }
    __syncthreads();

    const int b = blockIdx.x * 256 + t;
    const float4* xp = (const float4*)(x + (size_t)b * F_DIM);

    float ag[N_EXP], an[N_EXP];
    #pragma unroll
    for (int nn = 0; nn < N_EXP; ++nn) { ag[nn] = 0.f; an[nn] = 0.f; }

    for (int q = 0; q < F_DIM / 4; ++q) {
        float4 xv = xp[q];
        #pragma unroll
        for (int nn = 0; nn < N_EXP; ++nn) {
            float4 wgv = ((const float4*)wg)[nn * (F_DIM / 4) + q];
            float4 wnv = ((const float4*)wn)[nn * (F_DIM / 4) + q];
            ag[nn] += xv.x * wgv.x + xv.y * wgv.y + xv.z * wgv.z + xv.w * wgv.w;
            an[nn] += xv.x * wnv.x + xv.y * wnv.y + xv.z * wnv.z + xv.w * wnv.w;
        }
    }

    float th[N_EXP];
    float mx = -3.4e38f;
    #pragma unroll
    for (int nn = 0; nn < N_EXP; ++nn) {
        float g  = ag[nn] + Wg_b[nn];
        float v  = an[nn] + Wn_b[nn];
        float sp = (v > 20.f) ? v : log1pf(expf(v));     // stable softplus
        float h  = g + noise[nn] * sp;
        th[nn]   = (h <= 0.f) ? -1e38f : h;              // differentiable_threshold fwd
        mx = fmaxf(mx, th[nn]);
    }
    float e[N_EXP], s = 0.f;
    #pragma unroll
    for (int nn = 0; nn < N_EXP; ++nn) {
        e[nn] = expf(th[nn] - mx);                       // masked -> exp(-huge) == 0
        s += e[nn];
    }
    const float inv = 1.f / s;

    float G[N_EXP];
    int ksel = 0;
    #pragma unroll
    for (int nn = 0; nn < N_EXP; ++nn) {
        G[nn] = e[nn] * inv;
        ksel += (G[nn] > 0.f) ? 1 : 0;
    }
    int a = 0, c = ksel;
    #pragma unroll
    for (int nn = 0; nn < N_EXP; ++nn) {
        bool sel = (G[nn] > 0.f);
        int  sl  = sel ? a++ : c++;
        G_out  [(size_t)b * N_EXP + nn] = G[nn];
        gate_ws[(size_t)b * N_EXP + nn] = sel ? G[nn] : 0.f;
        slot_ws[(size_t)b * N_EXP + nn] = sl;
    }
}

// ---------------------------------------------------------------------------
// Kernel 2: per-expert GEMM via v_wmma_f32_16x16x32_bf16 + gated compaction.
// Block = (sample tile of 32, expert n).  LDS: x tile (bf16) + half of the
// expert's 128x256 weights (bf16).  8 waves each own one 16x16 output tile
// per half; K-loop = 8 WMMAs.  Fragments follow the CDNA5 16-bit A/B VGPR
// layouts and are 16B-contiguous in LDS (-> ds_load_b128).
// ---------------------------------------------------------------------------
__global__ void __launch_bounds__(256)
expert_kernel(const float* __restrict__ x,
              const float* __restrict__ comp_w,
              const float* __restrict__ comp_b,
              const float* __restrict__ gate_ws,
              const int*   __restrict__ slot_ws,
              float* __restrict__ out)
{
    __shared__ __align__(16) unsigned short xs[BM * F_DIM];   // 16 KB
    __shared__ __align__(16) unsigned short bw[64 * F_DIM];   // 32 KB

    const int n    = blockIdx.y;          // expert
    const int b0   = blockIdx.x * BM;     // sample base
    const int t    = threadIdx.x;
    const int wave = t >> 5;
    const int lane = t & 31;
    const int lr   = lane & 15;
    const int lh   = lane >> 4;
    const int mt   = wave & 1;            // M-tile 0..1 (16 samples each)
    const int dt   = wave >> 1;           // d-tile 0..3 within 64-col half

    // stage x tile, f32 -> bf16
    {
        const float4* src = (const float4*)(x + (size_t)b0 * F_DIM);
        uint2* dst = (uint2*)xs;
        for (int i = t; i < BM * F_DIM / 4; i += 256) {
            float4 v = src[i];
            dst[i] = make_uint2(pack2(v.x, v.y), pack2(v.z, v.w));
        }
    }

    for (int hd = 0; hd < 2; ++hd) {
        if (hd) __syncthreads();          // protect bw reuse
        {
            const float4* src =
                (const float4*)(comp_w + ((size_t)n * D_DIM + hd * 64) * F_DIM);
            uint2* dst = (uint2*)bw;
            for (int i = t; i < 64 * F_DIM / 4; i += 256) {
                float4 v = src[i];
                dst[i] = make_uint2(pack2(v.x, v.y), pack2(v.z, v.w));
            }
        }
        __syncthreads();

        v8f acc = {0.f, 0.f, 0.f, 0.f, 0.f, 0.f, 0.f, 0.f};
        const int arow = (mt * 16 + lr) * F_DIM;   // A: lane holds row M=lr
        const int bcol = (dt * 16 + lr) * F_DIM;   // B: lane holds col N=lr

        #pragma unroll
        for (int k0 = 0; k0 < F_DIM; k0 += 32) {
            Frag A, Bf;
            // A 16x32 bf16: lane half lh -> K = k0+8*lh+[0..7], k0+16+8*lh+[0..7]
            A.q[0]  = *(const uint4*)&xs[arow + k0 + 8 * lh];
            A.q[1]  = *(const uint4*)&xs[arow + k0 + 16 + 8 * lh];
            // B 32x16 bf16: lane half lh -> K = k0+8*lh+[0..7], k0+16+8*lh+[0..7]
            Bf.q[0] = *(const uint4*)&bw[bcol + k0 + 8 * lh];
            Bf.q[1] = *(const uint4*)&bw[bcol + k0 + 16 + 8 * lh];
            acc = __builtin_amdgcn_wmma_f32_16x16x32_bf16(
                false, A.v, false, Bf.v, (short)0, acc, false, false);
        }

        // Epilogue: gate * (Y + bias), compacted store (slot bijection).
        const int   col  = hd * 64 + dt * 16 + lr;   // D: N = lr
        const float bias = comp_b[n * D_DIM + col];
        #pragma unroll
        for (int r = 0; r < 8; ++r) {                // D: M = r + 8*lh
            int   b  = b0 + mt * 16 + r + 8 * lh;
            float gv = gate_ws[(size_t)b * N_EXP + n];
            int   sl = slot_ws[(size_t)b * N_EXP + n];
            out[(size_t)b * (N_EXP * D_DIM) + sl * D_DIM + col] =
                gv * (acc[r] + bias);
        }
    }
}

// ---------------------------------------------------------------------------
extern "C" void kernel_launch(void* const* d_in, const int* in_sizes, int n_in,
                              void* d_out, int out_size, void* d_ws, size_t ws_size,
                              hipStream_t stream)
{
    const float* x      = (const float*)d_in[0];
    const float* noise  = (const float*)d_in[1];
    const float* Wg_w   = (const float*)d_in[2];
    const float* Wg_b   = (const float*)d_in[3];
    const float* Wn_w   = (const float*)d_in[4];
    const float* Wn_b   = (const float*)d_in[5];
    const float* comp_w = (const float*)d_in[6];
    const float* comp_b = (const float*)d_in[7];

    const int B = in_sizes[0] / F_DIM;            // 16384

    float* out     = (float*)d_out;               // [B, N*D]
    float* G_out   = out + (size_t)B * (N_EXP * D_DIM);  // [B, N] (tuple out #2)
    float* gate_ws = (float*)d_ws;                            // [B, N] f32
    int*   slot_ws = (int*)((char*)d_ws + (size_t)B * N_EXP * sizeof(float));

    gate_kernel<<<B / 256, 256, 0, stream>>>(
        x, noise, Wg_w, Wg_b, Wn_w, Wn_b, G_out, gate_ws, slot_ws);

    dim3 grid(B / BM, N_EXP);
    expert_kernel<<<grid, 256, 0, stream>>>(
        x, comp_w, comp_b, gate_ws, slot_ws, out);
}
